// TransformerBlock_37812892074092
// MI455X (gfx1250) — matrix operand order repaired and checked by reading, not compiled, not run on hardware
//
#include <hip/hip_runtime.h>
#include <hip/hip_bf16.h>

#define D_MODEL 1024
#define N_HEADS 16
#define HEAD_DIM 64
#define D_FF 4096
#define SEQ 2048
#define BATCH 2
#define M_ROWS (BATCH * SEQ) /* 4096 */

typedef __attribute__((ext_vector_type(16))) __bf16 v16bf;
typedef __attribute__((ext_vector_type(8)))  float  v8f;
typedef __attribute__((ext_vector_type(8)))  __bf16 bh8;

#define BCAT16(lo, hi) __builtin_shufflevector((lo), (hi), 0,1,2,3,4,5,6,7,8,9,10,11,12,13,14,15)

// ---------------------------------------------------------------------------
// f32 -> bf16 conversion (weights, once per launch)
// ---------------------------------------------------------------------------
__global__ void cvt_f32_bf16(const float* __restrict__ in, __bf16* __restrict__ out, int n) {
    int i = blockIdx.x * blockDim.x + threadIdx.x;
    if (i < n) out[i] = (__bf16)in[i];
}

// ---------------------------------------------------------------------------
// LayerNorm over D=1024, one row per 256-thread block (8 wave32 waves)
// ---------------------------------------------------------------------------
__global__ __launch_bounds__(256) void layernorm_bf16(
    const float* __restrict__ x, const float* __restrict__ g,
    const float* __restrict__ b, __bf16* __restrict__ out) {
    const int row = blockIdx.x;
    const float* xr = x + (size_t)row * D_MODEL;
    float v[4];
    float s = 0.f;
#pragma unroll
    for (int j = 0; j < 4; ++j) { v[j] = xr[threadIdx.x + j * 256]; s += v[j]; }
    for (int o = 16; o > 0; o >>= 1) s += __shfl_xor(s, o, 32);
    __shared__ float red[8];
    const int wave = threadIdx.x >> 5, lane = threadIdx.x & 31;
    if (lane == 0) red[wave] = s;
    __syncthreads();
    float tot = 0.f;
#pragma unroll
    for (int w = 0; w < 8; ++w) tot += red[w];
    const float mu = tot * (1.0f / D_MODEL);
    float s2 = 0.f;
#pragma unroll
    for (int j = 0; j < 4; ++j) { float d = v[j] - mu; s2 += d * d; }
    for (int o = 16; o > 0; o >>= 1) s2 += __shfl_xor(s2, o, 32);
    __syncthreads();
    if (lane == 0) red[wave] = s2;
    __syncthreads();
    float tv = 0.f;
#pragma unroll
    for (int w = 0; w < 8; ++w) tv += red[w];
    const float rstd = rsqrtf(tv * (1.0f / D_MODEL) + 1e-5f);
#pragma unroll
    for (int j = 0; j < 4; ++j) {
        const int c = threadIdx.x + j * 256;
        out[(size_t)row * D_MODEL + c] = (__bf16)((v[j] - mu) * rstd * g[c] + b[c]);
    }
}

// ---------------------------------------------------------------------------
// bf16 WMMA GEMM:  Y[M,N] = act(X[M,K] @ W[N,K]^T + bias) (+ resid)
// 256 threads = 8 waves (2x4 grid); block tile 128x128; wave tile 64x32
// (4x2 WMMA subtiles); K stepped by 32 through LDS. Per K-step each wave
// issues 8 v_wmma against 12 ds_load_b128 (B frags reused over 4 M-subtiles).
// ---------------------------------------------------------------------------
__global__ __launch_bounds__(256) void gemm_wmma_bf16(
    const __bf16* __restrict__ A, const __bf16* __restrict__ W,
    const float* __restrict__ bias, const float* __restrict__ resid,
    float* __restrict__ outF, __bf16* __restrict__ outB,
    int M, int N, int K, int doGelu) {
    __shared__ __align__(16) __bf16 lA[128][40]; // 128 rows x 32 k-cols (padded)
    __shared__ __align__(16) __bf16 lB[128][40];
    const int tid = threadIdx.x;
    const int wave = tid >> 5, lane = tid & 31;
    const int wr = wave >> 2, wc = wave & 3;           // 2x4 wave grid
    const int tileM = blockIdx.y * 128, tileN = blockIdx.x * 128;
    const int ldRow = tid >> 1;                        // 0..127
    const int ldCol = (tid & 1) * 16;                  // 0 / 16
    const int mL = lane & 15, g = lane >> 4;           // WMMA lane decomposition

    v8f acc[4][2] = {{{}, {}}, {{}, {}}, {{}, {}}, {{}, {}}};
    const __bf16* aRow = A + (size_t)(tileM + ldRow) * K + ldCol;
    const __bf16* wRow = W + (size_t)(tileN + ldRow) * K + ldCol;

    for (int k0 = 0; k0 < K; k0 += 32) {
        bh8 a0 = *(const bh8*)(aRow + k0);
        bh8 a1 = *(const bh8*)(aRow + k0 + 8);
        bh8 w0 = *(const bh8*)(wRow + k0);
        bh8 w1 = *(const bh8*)(wRow + k0 + 8);
        if (k0 + 32 < K) {
            __builtin_prefetch(aRow + k0 + 32, 0, 3);   // global_prefetch_b8
            __builtin_prefetch(wRow + k0 + 32, 0, 3);
        }
        __syncthreads(); // previous tile fully consumed
        *(bh8*)&lA[ldRow][ldCol]     = a0;
        *(bh8*)&lA[ldRow][ldCol + 8] = a1;
        *(bh8*)&lB[ldRow][ldCol]     = w0;
        *(bh8*)&lB[ldRow][ldCol + 8] = w1;
        __syncthreads();
        // B fragments for this wave's two N-subtiles (reused over 4 M-subtiles)
        v16bf bf[2];
#pragma unroll
        for (int sj = 0; sj < 2; ++sj) {
            const int br = wc * 32 + sj * 16 + mL;      // lane = output column
            bh8 blo = *(const bh8*)&lB[br][8 * g];      // K = 8g..8g+7
            bh8 bhi = *(const bh8*)&lB[br][16 + 8 * g]; // K = 16+8g..
            bf[sj] = BCAT16(blo, bhi);
        }
#pragma unroll
        for (int si = 0; si < 4; ++si) {
            const int ar = wr * 64 + si * 16 + mL;
            bh8 alo = *(const bh8*)&lA[ar][8 * g];        // K = 8g..8g+7
            bh8 ahi = *(const bh8*)&lA[ar][16 + 8 * g];   // K = 16+8g..
            v16bf af = BCAT16(alo, ahi);
#pragma unroll
            for (int sj = 0; sj < 2; ++sj) {
                acc[si][sj] = __builtin_amdgcn_wmma_f32_16x16x32_bf16(
                    false, af, false, bf[sj], (short)0, acc[si][sj], false, false);
            }
        }
    }
    // Epilogue: bias (+gelu) (+resid); C layout: lane holds N=mL, rows r+8g.
#pragma unroll
    for (int si = 0; si < 4; ++si) {
#pragma unroll
        for (int sj = 0; sj < 2; ++sj) {
            const int col = tileN + wc * 32 + sj * 16 + mL;
            const float bia = bias ? bias[col] : 0.f;
#pragma unroll
            for (int r = 0; r < 8; ++r) {
                const int row = tileM + wr * 64 + si * 16 + r + 8 * g;
                float v = acc[si][sj][r] + bia;
                if (doGelu) v = 0.5f * v * (1.0f + erff(v * 0.70710678118654752f));
                if (resid) v += resid[(size_t)row * N + col];
                if (outF) outF[(size_t)row * N + col] = v;
                if (outB) outB[(size_t)row * N + col] = (__bf16)v;
            }
        }
    }
}

// ---------------------------------------------------------------------------
// Split fused QKV [M,3D] into Qh/Kh [B,H,S,hd] and Vt [B,H,hd,S] (transposed)
// ---------------------------------------------------------------------------
__global__ void qkv_split(const __bf16* __restrict__ qkv, __bf16* __restrict__ Qh,
                          __bf16* __restrict__ Kh, __bf16* __restrict__ Vt) {
    const int idx = blockIdx.x * blockDim.x + threadIdx.x;
    if (idx >= M_ROWS * D_MODEL) return;
    const int row = idx / D_MODEL;   // b*S + s
    const int d   = idx % D_MODEL;
    const int b = row / SEQ, s = row % SEQ;
    const int h = d >> 6, dd = d & 63;
    const __bf16* src = qkv + (size_t)row * (3 * D_MODEL);
    const size_t headBase = ((size_t)(b * N_HEADS + h) * SEQ + s) * HEAD_DIM + dd;
    Qh[headBase] = src[d];
    Kh[headBase] = src[D_MODEL + d];
    Vt[((size_t)(b * N_HEADS + h) * HEAD_DIM + dd) * SEQ + s] = src[2 * D_MODEL + d];
}

// ---------------------------------------------------------------------------
// Flash attention: one wave32 per 16-query tile; 32-key blocks; online softmax
// ---------------------------------------------------------------------------
__global__ __launch_bounds__(32) void attn_fwd(
    const __bf16* __restrict__ Qh, const __bf16* __restrict__ Kh,
    const __bf16* __restrict__ Vt, __bf16* __restrict__ out) {
    __shared__ __align__(16) __bf16 pLds[16][40];
    const int lane = threadIdx.x & 31;
    const int mL = lane & 15, g = lane >> 4;
    const int qbase = blockIdx.x * 16;
    const int h = blockIdx.y, b = blockIdx.z;
    const size_t headOff = (size_t)(b * N_HEADS + h) * SEQ * HEAD_DIM;

    // Q fragments (A layout), head_dim 64 = two K=32 chunks
    const __bf16* qptr = Qh + headOff + (size_t)(qbase + mL) * HEAD_DIM;
    v16bf aq[2];
#pragma unroll
    for (int c = 0; c < 2; ++c) {
        bh8 lo = *(const bh8*)(qptr + 32 * c + 8 * g);
        bh8 hi = *(const bh8*)(qptr + 32 * c + 16 + 8 * g);
        aq[c] = BCAT16(lo, hi);
    }
    v8f O[4] = {{}, {}, {}, {}};
    float mI[8], lI[8];
#pragma unroll
    for (int r = 0; r < 8; ++r) { mI[r] = -__builtin_inff(); lI[r] = 0.f; }
    const float scale = 0.125f; // 1/sqrt(64)
    const __bf16* kbaseptr = Kh + headOff;
    const __bf16* vbase = Vt + (size_t)(b * N_HEADS + h) * HEAD_DIM * SEQ;
    const int nkb = qbase / 32 + 1;

    for (int kb = 0; kb < nkb; ++kb) {
        const int k0 = kb * 32;
        v8f sc[2] = {{}, {}};
#pragma unroll
        for (int t = 0; t < 2; ++t) {
            const __bf16* kr = kbaseptr + (size_t)(k0 + 16 * t + mL) * HEAD_DIM;
#pragma unroll
            for (int c = 0; c < 2; ++c) {
                bh8 lo = *(const bh8*)(kr + 32 * c + 8 * g);
                bh8 hi = *(const bh8*)(kr + 32 * c + 16 + 8 * g);
                v16bf bk = BCAT16(lo, hi);
                sc[t] = __builtin_amdgcn_wmma_f32_16x16x32_bf16(
                    false, aq[c], false, bk, (short)0, sc[t], false, false);
            }
        }
        // causal mask + online softmax (rows live across the 16-lane half)
        float p0[8], p1[8];
#pragma unroll
        for (int r = 0; r < 8; ++r) {
            const int q = qbase + r + 8 * g;
            float v0 = sc[0][r] * scale; if (k0 + mL      > q) v0 = -__builtin_inff();
            float v1 = sc[1][r] * scale; if (k0 + 16 + mL > q) v1 = -__builtin_inff();
            float rm = fmaxf(v0, v1);
            for (int o = 8; o > 0; o >>= 1) rm = fmaxf(rm, __shfl_xor(rm, o, 32));
            const float nm = fmaxf(mI[r], rm);
            const float e0 = __expf(v0 - nm), e1 = __expf(v1 - nm);
            float rs = e0 + e1;
            for (int o = 8; o > 0; o >>= 1) rs += __shfl_xor(rs, o, 32);
            const float al = __expf(mI[r] - nm);
            lI[r] = lI[r] * al + rs;
            mI[r] = nm;
#pragma unroll
            for (int dt = 0; dt < 4; ++dt) O[dt][r] *= al;
            p0[r] = e0; p1[r] = e1;
        }
        // C-layout P -> LDS -> A-layout bf16 fragment
#pragma unroll
        for (int r = 0; r < 8; ++r) {
            pLds[r + 8 * g][mL]      = (__bf16)p0[r];
            pLds[r + 8 * g][16 + mL] = (__bf16)p1[r];
        }
        __syncthreads();
        bh8 plo = *(const bh8*)&pLds[mL][8 * g];
        bh8 phi = *(const bh8*)&pLds[mL][16 + 8 * g];
        v16bf ap = BCAT16(plo, phi);
#pragma unroll
        for (int dt = 0; dt < 4; ++dt) {
            const __bf16* vr = vbase + (size_t)(dt * 16 + mL) * SEQ + k0 + 8 * g;
            bh8 vlo = *(const bh8*)vr;
            bh8 vhi = *(const bh8*)(vr + 16);
            v16bf bv = BCAT16(vlo, vhi);
            O[dt] = __builtin_amdgcn_wmma_f32_16x16x32_bf16(
                false, ap, false, bv, (short)0, O[dt], false, false);
        }
        __syncthreads();
    }
    // normalize + write [B,S,D] bf16
#pragma unroll
    for (int dt = 0; dt < 4; ++dt) {
        const int col = h * HEAD_DIM + dt * 16 + mL;
#pragma unroll
        for (int r = 0; r < 8; ++r) {
            const int q = qbase + r + 8 * g;
            out[((size_t)(b * SEQ + q)) * D_MODEL + col] = (__bf16)(O[dt][r] / lI[r]);
        }
    }
}

// ---------------------------------------------------------------------------
// Host orchestration
// ---------------------------------------------------------------------------
extern "C" void kernel_launch(void* const* d_in, const int* in_sizes, int n_in,
                              void* d_out, int out_size, void* d_ws, size_t ws_size,
                              hipStream_t stream) {
    (void)in_sizes; (void)n_in; (void)out_size; (void)ws_size;
    const float* x    = (const float*)d_in[0];
    const float* ln1g = (const float*)d_in[1];
    const float* ln1b = (const float*)d_in[2];
    const float* qkvw = (const float*)d_in[3];
    const float* qkvb = (const float*)d_in[4];
    const float* outw = (const float*)d_in[5];
    const float* outb = (const float*)d_in[6];
    const float* ln2g = (const float*)d_in[7];
    const float* ln2b = (const float*)d_in[8];
    const float* fc1w = (const float*)d_in[9];
    const float* fc1b = (const float*)d_in[10];
    const float* fc2w = (const float*)d_in[11];
    const float* fc2b = (const float*)d_in[12];
    float* outp = (float*)d_out;

    char* ws = (char*)d_ws;
    size_t off = 0;
    auto alloc = [&](size_t bytes) -> char* {
        char* p = ws + off;
        off = (off + bytes + 255) & ~(size_t)255;
        return p;
    };
    __bf16* wb_qkv = (__bf16*)alloc((size_t)3 * D_MODEL * D_MODEL * 2);
    __bf16* wb_out = (__bf16*)alloc((size_t)D_MODEL * D_MODEL * 2);
    __bf16* wb_fc1 = (__bf16*)alloc((size_t)D_FF * D_MODEL * 2);
    __bf16* wb_fc2 = (__bf16*)alloc((size_t)D_MODEL * D_FF * 2);
    __bf16* h1     = (__bf16*)alloc((size_t)M_ROWS * D_MODEL * 2);
    __bf16* qkv    = (__bf16*)alloc((size_t)M_ROWS * 3 * D_MODEL * 2);
    __bf16* Qh     = (__bf16*)alloc((size_t)M_ROWS * D_MODEL * 2);
    __bf16* Kh     = (__bf16*)alloc((size_t)M_ROWS * D_MODEL * 2);
    __bf16* Vt     = (__bf16*)alloc((size_t)M_ROWS * D_MODEL * 2);
    __bf16* attnb  = (__bf16*)alloc((size_t)M_ROWS * D_MODEL * 2);
    float*  x2     = (float*)alloc((size_t)M_ROWS * D_MODEL * 4);
    __bf16* h2     = (__bf16*)alloc((size_t)M_ROWS * D_MODEL * 2);
    __bf16* ff     = (__bf16*)alloc((size_t)M_ROWS * D_FF * 2);

    int n;
    n = 3 * D_MODEL * D_MODEL;
    cvt_f32_bf16<<<(n + 255) / 256, 256, 0, stream>>>(qkvw, wb_qkv, n);
    n = D_MODEL * D_MODEL;
    cvt_f32_bf16<<<(n + 255) / 256, 256, 0, stream>>>(outw, wb_out, n);
    n = D_FF * D_MODEL;
    cvt_f32_bf16<<<(n + 255) / 256, 256, 0, stream>>>(fc1w, wb_fc1, n);
    n = D_MODEL * D_FF;
    cvt_f32_bf16<<<(n + 255) / 256, 256, 0, stream>>>(fc2w, wb_fc2, n);

    // LN1
    layernorm_bf16<<<M_ROWS, 256, 0, stream>>>(x, ln1g, ln1b, h1);
    // QKV projection: [4096,1024] x [3072,1024]^T
    gemm_wmma_bf16<<<dim3(3 * D_MODEL / 128, M_ROWS / 128), 256, 0, stream>>>(
        h1, wb_qkv, qkvb, nullptr, nullptr, qkv, M_ROWS, 3 * D_MODEL, D_MODEL, 0);
    n = M_ROWS * D_MODEL;
    qkv_split<<<(n + 255) / 256, 256, 0, stream>>>(qkv, Qh, Kh, Vt);
    // attention: 128 q-tiles x 16 heads x 2 batch, 1 wave each
    attn_fwd<<<dim3(SEQ / 16, N_HEADS, BATCH), 32, 0, stream>>>(Qh, Kh, Vt, attnb);
    // out projection + residual(x) -> x2 (f32)
    gemm_wmma_bf16<<<dim3(D_MODEL / 128, M_ROWS / 128), 256, 0, stream>>>(
        attnb, wb_out, outb, x, x2, nullptr, M_ROWS, D_MODEL, D_MODEL, 0);
    // LN2
    layernorm_bf16<<<M_ROWS, 256, 0, stream>>>(x2, ln2g, ln2b, h2);
    // FC1 + exact GELU
    gemm_wmma_bf16<<<dim3(D_FF / 128, M_ROWS / 128), 256, 0, stream>>>(
        h2, wb_fc1, fc1b, nullptr, nullptr, ff, M_ROWS, D_FF, D_MODEL, 1);
    // FC2 + residual(x2) -> out (f32)
    gemm_wmma_bf16<<<dim3(D_MODEL / 128, M_ROWS / 128), 256, 0, stream>>>(
        ff, wb_fc2, fc2b, x2, outp, nullptr, M_ROWS, D_MODEL, D_FF, 0);
}
